// MPLayer_5677946765362
// MI455X (gfx1250) — matrix operand order repaired and checked by reading, not compiled, not run on hardware
//
#include <hip/hip_runtime.h>
#include <hip/hip_bf16.h>

// ---------------------------------------------------------------------------
// MP layer, fused for gfx1250 (wave32, WMMA f32_16x16x32_f16).
//   B=64, N=100, D=32 ; edge MLP 64->96->160->192 (lrelu all) ;
//   sum over j ; node MLP 224->256->256->32 (last linear).
// Layer0 is decomposed: pre0(i,j) = P[i] + Q[j],  P = x@W0_top + b0, Q = x@W0_bot.
// ---------------------------------------------------------------------------

typedef __attribute__((ext_vector_type(16))) _Float16 v16h;
typedef __attribute__((ext_vector_type(8)))  _Float16 v8h;
typedef __attribute__((ext_vector_type(8)))  float    v8f;
typedef __attribute__((ext_vector_type(4)))  float    v4f;

#define MP_B    64
#define MP_N    100
#define MP_D    32
#define MP_IT   7            // ceil(100/16) i-tiles
#define ALPHA_LR 0.2f

__device__ __forceinline__ float lrelu(float v) { return fmaxf(v, ALPHA_LR * v); }

// ---------------------------------------------------------------------------
// prep: P[bn][c] = sum_d x[bn][d]*W0[d][c] + b0[c];  Q uses W0 rows 32..63
// ---------------------------------------------------------------------------
__global__ void mp_prep_pq(const float* __restrict__ x, const float* __restrict__ w0,
                           const float* __restrict__ b0,
                           float* __restrict__ P, float* __restrict__ Q) {
    int idx = blockIdx.x * blockDim.x + threadIdx.x;     // over (B*N)*96
    if (idx >= MP_B * MP_N * 96) return;
    int bn = idx / 96, c = idx % 96;
    const float* xr = x + bn * MP_D;
    float p = b0[c], q = 0.f;
#pragma unroll
    for (int d = 0; d < MP_D; ++d) {
        p += xr[d] * w0[d * 96 + c];
        q += xr[d] * w0[(MP_D + d) * 96 + c];
    }
    P[idx] = p; Q[idx] = q;
}

// ---------------------------------------------------------------------------
// Pack a row-major [K][Nout] f32 weight into WMMA B-operand tiles (f16).
// Tile (kt,nt) = 512 halves; lane l holds col n = nt*16 + l%16 and
// K = kt*32 + (l/16)*8 + (h/8)*16 + h%8  for h = 0..15 (contiguous 32B per lane).
// ---------------------------------------------------------------------------
__global__ void mp_pack_b(const float* __restrict__ W, int K, int Nout,
                          _Float16* __restrict__ out) {
    int idx = blockIdx.x * blockDim.x + threadIdx.x;
    if (idx >= K * Nout) return;
    int ntiles = Nout >> 4;
    int tile = idx >> 9;
    int rem  = idx & 511;
    int lane = rem >> 4;
    int h    = rem & 15;
    int kt = tile / ntiles, nt = tile % ntiles;
    int n = nt * 16 + (lane & 15);
    int k = kt * 32 + (lane >> 4) * 8 + (h >> 3) * 16 + (h & 7);
    out[idx] = (_Float16)W[k * Nout + n];
}

// ---------------------------------------------------------------------------
// Generic WMMA n-tile: C(16x16) = bias + A(16xK,from LDS row-major halves) x B
// A-operand layout: lanes 0-15 row m=l, K runs 0..7 / 16..23 ; lanes 16-31
// runs 8..15 / 24..31  (ISA 7.12.2, 16-bit A 16x32).
// ---------------------------------------------------------------------------
__device__ __forceinline__ v8f mp_gemm_tile(const _Float16* Ain, int Aw,
                                            const _Float16* Bp, int ntiles,
                                            float bias, int n, int nkt, int lane) {
    v8f c;
#pragma unroll
    for (int e = 0; e < 8; ++e) c[e] = bias;
    int m = lane & 15;
    int koffA = (lane >> 4) * 8;
    for (int kt = 0; kt < nkt; ++kt) {
        const _Float16* sp = Ain + m * Aw + kt * 32 + koffA;
        v8h lo = *(const v8h*)sp;
        v8h hi = *(const v8h*)(sp + 16);
        v16h a = __builtin_shufflevector(lo, hi, 0,1,2,3,4,5,6,7,8,9,10,11,12,13,14,15);
        v16h bt = *(const v16h*)(Bp + (size_t)(kt * ntiles + n) * 512 + lane * 16);
        c = __builtin_amdgcn_wmma_f32_16x16x32_f16(false, a, false, bt, (short)0, c,
                                                   false, false);
    }
    return c;
}

// ---------------------------------------------------------------------------
// Fused main kernel. Grid = B*7 WGs, 128 threads (4 wave32).
// LDS (dynamic, 126336 B):
//   [     0, 30720) W1 tiles (96x160 f16)      [ 30720, 92160) W2 (160x192 f16)
//   [ 92160,112640) per-wave stage 16x160 f16  [112640,124928) agg 16x192 f32
//   [124928,126336) edge biases b1(160)+b2(192) f32
// Node phase aliases: hstage@92160 (16x224), n0out@99328 (16x256),
//                     n1out@112640 (16x256, reuses agg after it is consumed).
// amdgpu_waves_per_eu(4): cap VGPRs at 256/wave -> no s_set_vgpr_msb in the
// hot loop and a smaller wave footprint.
// ---------------------------------------------------------------------------
__global__ __launch_bounds__(128)
__attribute__((amdgpu_waves_per_eu(4)))
void mp_main(
    const float* __restrict__ x,
    const float* __restrict__ P, const float* __restrict__ Q,
    const _Float16* __restrict__ w1p, const float* __restrict__ b1,
    const _Float16* __restrict__ w2p, const float* __restrict__ b2,
    const _Float16* __restrict__ wn0p, const float* __restrict__ nb0,
    const _Float16* __restrict__ wn1p, const float* __restrict__ nb1,
    const _Float16* __restrict__ wn2p, const float* __restrict__ nb2,
    float* __restrict__ out) {
    extern __shared__ char smem[];
    _Float16* sW1 = (_Float16*)smem;                 // [3][10][512]
    _Float16* sW2 = (_Float16*)(smem + 30720);       // [5][12][512]
    float*    aggS = (float*)(smem + 112640);        // [16][192]
    float*    sB1  = (float*)(smem + 124928);        // [160]
    float*    sB2  = sB1 + 160;                      // [192]

    const int tid  = threadIdx.x;
    const int wave = tid >> 5;
    const int lane = tid & 31;
    const int r0    = (lane >> 4) * 8;               // C-matrix row base
    const int koffA = (lane >> 4) * 8;               // A-operand K-run base
    const int cn    = lane & 15;                     // C-matrix column / A row

    _Float16* stage = (_Float16*)(smem + 92160) + wave * (16 * 160);

    const int b  = blockIdx.x / MP_IT;
    const int it = blockIdx.x % MP_IT;

    // --- stage edge weights + biases into LDS, zero agg --------------------
    {
        const uint4* s1 = (const uint4*)w1p;
        const uint4* s2 = (const uint4*)w2p;
        uint4* d1 = (uint4*)smem;
        uint4* d2 = (uint4*)(smem + 30720);
        for (int i = tid; i < 30720 / 16; i += 128) d1[i] = s1[i];
        for (int i = tid; i < 61440 / 16; i += 128) d2[i] = s2[i];
        for (int i = tid; i < 16 * 192; i += 128) aggS[i] = 0.f;
        for (int i = tid; i < 160; i += 128) sB1[i] = b1[i];
        for (int i = tid; i < 192; i += 128) sB2[i] = b2[i];
    }
    __syncthreads();

    // =======================================================================
    // Edge phase: 112 M-tiles (16 i-locals x 7 j-tiles), 4 waves round-robin.
    // =======================================================================
    for (int t = wave; t < 16 * MP_IT; t += 4) {
        const int il = t / MP_IT;            // i within tile
        const int jt = t % MP_IT;            // j tile
        const int ig = min(it * 16 + il, MP_N - 1);
        const int jg = min(jt * 16 + cn, MP_N - 1);
        const float* pr = P + (size_t)(b * MP_N + ig) * 96;
        const float* qr = Q + (size_t)(b * MP_N + jg) * 96;

        // ---- build A0 = lrelu(P[i] + Q[j]) directly in A-operand layout ----
        v16h a0[3];
#pragma unroll
        for (int kt = 0; kt < 3; ++kt) {
#pragma unroll
            for (int r2 = 0; r2 < 2; ++r2) {
                int ko = kt * 32 + koffA + r2 * 16;
                v4f p0 = *(const v4f*)(pr + ko);
                v4f p1 = *(const v4f*)(pr + ko + 4);
                v4f q0 = *(const v4f*)(qr + ko);
                v4f q1 = *(const v4f*)(qr + ko + 4);
#pragma unroll
                for (int e = 0; e < 4; ++e) {
                    a0[kt][r2 * 8 + e]     = (_Float16)lrelu(p0[e] + q0[e]);
                    a0[kt][r2 * 8 + 4 + e] = (_Float16)lrelu(p1[e] + q1[e]);
                }
            }
        }

        // ---- layer 1: 16x96 @ 96x160, stage lrelu result (f16) in LDS ------
#pragma unroll
        for (int n1 = 0; n1 < 10; ++n1) {
            float bias = sB1[n1 * 16 + cn];
            v8f c;
#pragma unroll
            for (int e = 0; e < 8; ++e) c[e] = bias;
#pragma unroll
            for (int kt = 0; kt < 3; ++kt) {
                v16h bt = *(const v16h*)(sW1 + (kt * 10 + n1) * 512 + lane * 16);
                c = __builtin_amdgcn_wmma_f32_16x16x32_f16(false, a0[kt], false, bt,
                                                           (short)0, c, false, false);
            }
            int col = n1 * 16 + cn;
#pragma unroll
            for (int r = 0; r < 8; ++r)
                stage[(r0 + r) * 160 + col] = (_Float16)lrelu(c[r]);
        }

        // ---- reload as A-operand (cross-lane transpose through LDS) --------
        v16h a1[5];
#pragma unroll
        for (int kt = 0; kt < 5; ++kt) {
            const _Float16* sp = stage + cn * 160 + kt * 32 + koffA;
            v8h lo = *(const v8h*)sp;
            v8h hi = *(const v8h*)(sp + 16);
            a1[kt] = __builtin_shufflevector(lo, hi, 0,1,2,3,4,5,6,7,8,9,10,11,12,13,14,15);
        }

        // ---- layer 2: 16x160 @ 160x192, mask pad rows, reduce over j -------
        const int jb = jt * 16 + r0;
#pragma unroll
        for (int n2 = 0; n2 < 12; ++n2) {
            float bias = sB2[n2 * 16 + cn];
            v8f c;
#pragma unroll
            for (int e = 0; e < 8; ++e) c[e] = bias;
#pragma unroll
            for (int kt = 0; kt < 5; ++kt) {
                v16h bt = *(const v16h*)(sW2 + (kt * 12 + n2) * 512 + lane * 16);
                c = __builtin_amdgcn_wmma_f32_16x16x32_f16(false, a1[kt], false, bt,
                                                           (short)0, c, false, false);
            }
            float s = 0.f;
#pragma unroll
            for (int r = 0; r < 8; ++r)
                if (jb + r < MP_N) s += lrelu(c[r]);
            atomicAdd(&aggS[il * 192 + n2 * 16 + cn], s);   // ds_add_f32
        }
    }
    __syncthreads();

    // =======================================================================
    // Node phase: 16 i-rows, MLP 224 -> 256 -> 256 -> 32.
    // =======================================================================
    _Float16* hstage = (_Float16*)(smem + 92160);            // [16][224]
    _Float16* n0out  = (_Float16*)(smem + 92160 + 7168);     // [16][256]
    _Float16* n1out  = (_Float16*)(smem + 112640);           // [16][256] (ex-agg)

    for (int i = tid; i < 16 * 224; i += 128) {
        int row = i / 224, col = i % 224;
        float v;
        if (col < 192) v = aggS[row * 192 + col];
        else {
            int ig = min(it * 16 + row, MP_N - 1);
            v = x[(size_t)(b * MP_N + ig) * MP_D + (col - 192)];
        }
        hstage[i] = (_Float16)v;
    }
    __syncthreads();

#pragma unroll
    for (int s = 0; s < 4; ++s) {                            // fn layer 0
        int n = wave * 4 + s;
        v8f c = mp_gemm_tile(hstage, 224, wn0p, 16, nb0[n * 16 + cn], n, 7, lane);
        int col = n * 16 + cn;
#pragma unroll
        for (int r = 0; r < 8; ++r)
            n0out[(r0 + r) * 256 + col] = (_Float16)lrelu(c[r]);
    }
    __syncthreads();

#pragma unroll
    for (int s = 0; s < 4; ++s) {                            // fn layer 1
        int n = wave * 4 + s;
        v8f c = mp_gemm_tile(n0out, 256, wn1p, 16, nb1[n * 16 + cn], n, 8, lane);
        int col = n * 16 + cn;
#pragma unroll
        for (int r = 0; r < 8; ++r)
            n1out[(r0 + r) * 256 + col] = (_Float16)lrelu(c[r]);
    }
    __syncthreads();

    if (wave < 2) {                                          // fn layer 2 (linear)
        int n = wave;
        v8f c = mp_gemm_tile(n1out, 256, wn2p, 2, nb2[n * 16 + cn], n, 8, lane);
#pragma unroll
        for (int r = 0; r < 8; ++r) {
            int igr = it * 16 + r0 + r;
            if (igr < MP_N)
                out[(size_t)(b * MP_N + igr) * 32 + n * 16 + cn] = c[r];
        }
    }
}

// ---------------------------------------------------------------------------
extern "C" void kernel_launch(void* const* d_in, const int* in_sizes, int n_in,
                              void* d_out, int out_size, void* d_ws, size_t ws_size,
                              hipStream_t stream) {
    const float* x    = (const float*)d_in[0];
    const float* few0 = (const float*)d_in[1];
    const float* feb0 = (const float*)d_in[2];
    const float* few1 = (const float*)d_in[3];
    const float* feb1 = (const float*)d_in[4];
    const float* few2 = (const float*)d_in[5];
    const float* feb2 = (const float*)d_in[6];
    const float* fnw0 = (const float*)d_in[7];
    const float* fnb0 = (const float*)d_in[8];
    const float* fnw1 = (const float*)d_in[9];
    const float* fnb1 = (const float*)d_in[10];
    const float* fnw2 = (const float*)d_in[11];
    const float* fnb2 = (const float*)d_in[12];

    char* ws = (char*)d_ws;
    float* P = (float*)ws;                              // 6400*96 f32
    float* Q = P + MP_B * MP_N * 96;                    // 6400*96 f32
    _Float16* w1p  = (_Float16*)(Q + MP_B * MP_N * 96); // 96*160
    _Float16* w2p  = w1p + 96 * 160;                    // 160*192
    _Float16* wn0p = w2p + 160 * 192;                   // 224*256
    _Float16* wn1p = wn0p + 224 * 256;                  // 256*256
    _Float16* wn2p = wn1p + 256 * 256;                  // 256*32

    mp_prep_pq<<<(MP_B * MP_N * 96 + 255) / 256, 256, 0, stream>>>(x, few0, feb0, P, Q);
    mp_pack_b<<<(96 * 160 + 255) / 256, 256, 0, stream>>>(few1, 96, 160, w1p);
    mp_pack_b<<<(160 * 192 + 255) / 256, 256, 0, stream>>>(few2, 160, 192, w2p);
    mp_pack_b<<<(224 * 256 + 255) / 256, 256, 0, stream>>>(fnw0, 224, 256, wn0p);
    mp_pack_b<<<(256 * 256 + 255) / 256, 256, 0, stream>>>(fnw1, 256, 256, wn1p);
    mp_pack_b<<<(256 * 32 + 255) / 256, 256, 0, stream>>>(fnw2, 256, 32, wn2p);

    const int shmem = 126336;   // > 64KB: fine on CDNA5 (320KB/WGP), opt in
    (void)hipFuncSetAttribute((const void*)mp_main,
                              hipFuncAttributeMaxDynamicSharedMemorySize, shmem);
    mp_main<<<MP_B * MP_IT, 128, shmem, stream>>>(
        x, P, Q, w1p, feb1, w2p, feb2, wn0p, fnb0, wn1p, fnb1, wn2p, fnb2,
        (float*)d_out);
}
